// GCNlayer_61065845015423
// MI455X (gfx1250) — matrix operand order repaired and checked by reading, not compiled, not run on hardware
//
#include <hip/hip_runtime.h>
#include <hip/hip_bf16.h>

#define N_NODES 100000
#define N_EDGES 3200000
#define IN_DIM 256
#define OUT_DIM 256
#define M_TILES (N_NODES / 16)     // 6250, exact
#define LDS_STRIDE 264             // 256 + 8 bf16 pad -> 4-bank rotation per column
#define LDS_BYTES (OUT_DIM * LDS_STRIDE * 2)   // 135168 B

typedef __bf16 v16bf __attribute__((ext_vector_type(16)));
typedef __bf16 v8bf  __attribute__((ext_vector_type(8)));
typedef float  v8f   __attribute__((ext_vector_type(8)));
typedef float  v4f   __attribute__((ext_vector_type(4)));
typedef int    v4i   __attribute__((ext_vector_type(4)));

#define AS1 __attribute__((address_space(1)))
#define AS3 __attribute__((address_space(3)))

#if defined(__has_builtin)
#if __has_builtin(__builtin_amdgcn_global_load_async_to_lds_b128)
#define HAVE_ASYNC_LDS 1
#endif
#endif

// ---- kernel 0: weight fp32 -> bf16, TRANSPOSED: wbt[n*256 + k] = w[k*256 + n] ----
__global__ __launch_bounds__(256) void gcn_wconv_t(const float* __restrict__ w,
                                                   __bf16* __restrict__ wbt) {
    int idx = blockIdx.x * 256 + threadIdx.x;   // 65536 elems, coalesced read
    int k = idx >> 8, n = idx & 255;
    wbt[n * IN_DIM + k] = (__bf16)w[idx];
}

// ---- kernel 1: h = x @ W via v_wmma_f32_16x16x32_bf16, weight staged in LDS ----
// block = 256 threads = 8 waves; covers 64 rows x 256 cols.
// wave w: M-tile = blk*4 + (w>>1); N-half = (w&1)*128 -> 8 accumulator tiles.
__global__ __launch_bounds__(256) void gcn_gemm(const float* __restrict__ x,
                                                const __bf16* __restrict__ wbt,
                                                float* __restrict__ h) {
    extern __shared__ __align__(16) char smem_raw[];
    __bf16* lwb = (__bf16*)smem_raw;           // [256 cols][LDS_STRIDE k]

    // ---- cooperative async copy of transposed weight into LDS ----
    // 256 rows x 32 chunks of 8 bf16 (16 B) = 8192 chunks, 32 per thread
    for (int c = threadIdx.x; c < 256 * 32; c += 256) {
        const int n  = c >> 5;
        const int c8 = (c & 31) * 8;
        const __bf16* gsrc = wbt + n * IN_DIM + c8;
        __bf16*       ldst = lwb + n * LDS_STRIDE + c8;
#ifdef HAVE_ASYNC_LDS
        __builtin_amdgcn_global_load_async_to_lds_b128(
            (AS1 v4i*)gsrc, (AS3 v4i*)ldst, 0, 0);
#else
        *(v8bf*)ldst = *(const v8bf*)gsrc;
#endif
    }
#ifdef HAVE_ASYNC_LDS
    asm volatile("s_wait_asynccnt 0" ::: "memory");
#endif
    __syncthreads();

    const int wave  = threadIdx.x >> 5;
    const int lane  = threadIdx.x & 31;
    const int mtile = blockIdx.x * 4 + (wave >> 1);
    if (mtile >= M_TILES) return;              // wave-uniform: EXEC stays all-ones
    const int nhalf = (wave & 1) * 128;
    const int row   = mtile * 16 + (lane & 15);
    const int kb    = (lane >> 4) * 8;         // A: K sub-offset per lane half
    const int khalf = (lane >> 4) * 8;         // B: K sub-offset per lane half
    const int ncol  = lane & 15;               // B/C/D column within tile

    v8f acc[8];
#pragma unroll
    for (int t = 0; t < 8; ++t) acc[t] = {};

    for (int ks = 0; ks < IN_DIM; ks += 32) {
        // --- A fragment: 16x32 bf16, converted from fp32 on the fly ---
        const float* xr = x + row * IN_DIM + ks + kb;
        v4f a0 = *(const v4f*)(xr);            // K = ks+kb+0..3
        v4f a1 = *(const v4f*)(xr + 4);        // K = ks+kb+4..7
        v4f a2 = *(const v4f*)(xr + 16);       // K = ks+kb+16..19
        v4f a3 = *(const v4f*)(xr + 20);       // K = ks+kb+20..23
        v16bf afrag;
#pragma unroll
        for (int j = 0; j < 4; ++j) {
            afrag[j]      = (__bf16)a0[j];
            afrag[4 + j]  = (__bf16)a1[j];
            afrag[8 + j]  = (__bf16)a2[j];
            afrag[12 + j] = (__bf16)a3[j];
        }
        // --- batch-issue all 16 ds_load_b128 for this k-step, then drain ---
        v8bf bl[8], bh[8];
#pragma unroll
        for (int t = 0; t < 8; ++t) {
            const __bf16* bp = lwb + (nhalf + t * 16 + ncol) * LDS_STRIDE + ks + khalf;
            bl[t] = *(const v8bf*)bp;
            bh[t] = *(const v8bf*)(bp + 16);
        }
#pragma unroll
        for (int t = 0; t < 8; ++t) {
            v16bf bfrag = __builtin_shufflevector(bl[t], bh[t],
                0, 1, 2, 3, 4, 5, 6, 7, 8, 9, 10, 11, 12, 13, 14, 15);
            acc[t] = __builtin_amdgcn_wmma_f32_16x16x32_bf16(
                false, afrag, false, bfrag, (short)0, acc[t], false, false);
        }
    }

    // --- store C tiles (16x16 f32 layout: reg r -> row r (+8 for hi half)) ---
    const int mo = (lane >> 4) * 8;
#pragma unroll
    for (int t = 0; t < 8; ++t) {
        float* hp = h + (mtile * 16 + mo) * OUT_DIM + nhalf + t * 16 + ncol;
#pragma unroll
        for (int r = 0; r < 8; ++r) hp[r * OUT_DIM] = acc[t][r];
    }
}

// ---- kernel 2: out[n, j] = bias[j] ----
__global__ __launch_bounds__(256) void gcn_init(const float* __restrict__ bias,
                                                float* __restrict__ out) {
    int idx = blockIdx.x * 256 + threadIdx.x;
    if (idx < N_NODES * OUT_DIM) out[idx] = bias[idx & (OUT_DIM - 1)];
}

// ---- kernel 3: out[dst] += val * h[src]   (one edge per wave iteration) ----
__global__ __launch_bounds__(256) void gcn_agg(const float* __restrict__ h,
                                               const int* __restrict__ esrc,
                                               const int* __restrict__ edst,
                                               const float* __restrict__ eval,
                                               float* __restrict__ out,
                                               int total_waves) {
    const int gwave = (blockIdx.x * blockDim.x + threadIdx.x) >> 5;
    const int lane  = threadIdx.x & 31;
    for (int e = gwave; e < N_EDGES; e += total_waves) {
        const int   s = esrc[e];
        const int   d = edst[e];
        const float v = eval[e];
        const float* hp = h   + (size_t)s * OUT_DIM + lane;
        float*       op = out + (size_t)d * OUT_DIM + lane;
#pragma unroll
        for (int j = 0; j < 8; ++j) {
            float m = v * hp[j * 32];   // 32 lanes x 4B contiguous per j
            __hip_atomic_fetch_add(op + j * 32, m,
                                   __ATOMIC_RELAXED, __HIP_MEMORY_SCOPE_AGENT);
        }
    }
}

extern "C" void kernel_launch(void* const* d_in, const int* in_sizes, int n_in,
                              void* d_out, int out_size, void* d_ws, size_t ws_size,
                              hipStream_t stream) {
    const float* x      = (const float*)d_in[0];
    const int*   esrc   = (const int*)  d_in[1];
    const int*   edst   = (const int*)  d_in[2];
    const float* eval   = (const float*)d_in[3];
    const float* weight = (const float*)d_in[4];
    const float* bias   = (const float*)d_in[5];
    float*       out    = (float*)d_out;

    // ws layout: [0, 128KB) transposed bf16 weight; [128KB, ...) h fp32 (102.4 MB)
    __bf16* wbt = (__bf16*)d_ws;
    float*  h   = (float*)((char*)d_ws + 131072);

    // 1) weight -> bf16, transposed to [out][in]
    gcn_wconv_t<<<IN_DIM * OUT_DIM / 256, 256, 0, stream>>>(weight, wbt);

    // 2) h = x @ W  (WMMA, LDS-staged weight); 4 M-tiles per block
    int gemm_blocks = (M_TILES + 3) / 4;   // 1563
    gcn_gemm<<<gemm_blocks, 256, LDS_BYTES, stream>>>(x, wbt, h);

    // 3) out = bias (broadcast)
    gcn_init<<<(N_NODES * OUT_DIM + 255) / 256, 256, 0, stream>>>(bias, out);

    // 4) edge aggregation with f32 atomics
    const int agg_blocks = 4096;
    const int total_waves = agg_blocks * (256 / 32);
    gcn_agg<<<agg_blocks, 256, 0, stream>>>(h, esrc, edst, eval, out, total_waves);
}
